// Attention_3161095929939
// MI455X (gfx1250) — matrix operand order repaired and checked by reading, not compiled, not run on hardware
//
#include <hip/hip_runtime.h>

// ---------------------------------------------------------------------------
// Types for CDNA5 WMMA (wave32): v_wmma_f32_16x16x32_bf16
// ---------------------------------------------------------------------------
typedef __attribute__((ext_vector_type(16))) __bf16 v16bf;
typedef __attribute__((ext_vector_type(8)))  float  v8f;
typedef __attribute__((ext_vector_type(4)))  unsigned int v4u;
typedef __attribute__((ext_vector_type(8)))  int    v8i;
typedef __attribute__((ext_vector_type(4)))  int    v4i;

union frag16 {            // 16 bf16 = 32 bytes = 8 VGPRs
    v16bf v;
    uint4 u[2];
};

__device__ inline v8f wmma_bf16(v16bf a, v16bf b, v8f c) {
    // (neg_a, A, neg_b, B, c_mod, C, reuse_a, reuse_b)
    return __builtin_amdgcn_wmma_f32_16x16x32_bf16(false, a, false, b,
                                                   (short)0, c, false, false);
}

__device__ inline __bf16 f2bf(float f) {
    union { float f; unsigned u; } a; a.f = f;
    unsigned r = (a.u + 0x7FFFu + ((a.u >> 16) & 1u)) >> 16;
    union { unsigned short s; __bf16 b; } o; o.s = (unsigned short)r;
    return o.b;
}

#define DIM      1024
#define NSEQ     2048
#define NB       2
#define HEADS    16
#define DH       64
#define MROWS    4096            // NB * NSEQ
#define SCALE    0.125f          // 64^-0.5

// ---------------------------------------------------------------------------
// TDM: 2-D tensor_load_to_lds (D# per CDNA5 ISA ch.8); 6-arg builtin variant
// ---------------------------------------------------------------------------
#if defined(__has_builtin)
#if __has_builtin(__builtin_amdgcn_tensor_load_to_lds) && \
    __has_builtin(__builtin_amdgcn_s_wait_tensorcnt)
#define HAVE_TDM 1
#endif
#endif

#ifdef HAVE_TDM
__device__ inline void tdm_load_2d(const void* gsrc, void* lds,
                                   unsigned tensor_d0, unsigned tensor_d1,
                                   unsigned tile_d0, unsigned tile_d1,
                                   unsigned long long stride0_elems) {
    unsigned long long ga = (unsigned long long)(uintptr_t)gsrc;
    unsigned la = (unsigned)(uintptr_t)lds;          // LDS byte address (addr[31:0])
    v4u g0;
    g0[0] = 1u;                                      // count=1 valid descriptor
    g0[1] = la;                                      // lds_addr [63:32]
    g0[2] = (unsigned)(ga & 0xFFFFFFFFu);            // global_addr lo  [95:64]
    g0[3] = (unsigned)((ga >> 32) & 0x1FFFFFFu)      // global_addr hi  [120:96]
          | (2u << 30);                              // type=2 ("image") [127:126]
    v8i g1;
    g1[0] = (int)(1u << 16);                         // data_size=1 (2 bytes)
    g1[1] = (int)((tensor_d0 & 0xFFFFu) << 16);      // tensor_dim0 lo16 @[63:48]
    g1[2] = (int)((tensor_d0 >> 16) | ((tensor_d1 & 0xFFFFu) << 16));
    g1[3] = (int)((tensor_d1 >> 16) | (tile_d0 << 16));     // tile_dim0 @[127:112]
    g1[4] = (int)(tile_d1 & 0xFFFFu);                // tile_dim1; tile_dim2=0 (2D)
    g1[5] = (int)(stride0_elems & 0xFFFFFFFFull);    // tensor_dim0_stride lo32
    g1[6] = (int)((stride0_elems >> 32) & 0xFFFFull);// stride0 hi16; stride1=0
    g1[7] = 0;
    v4i gz4 = {0, 0, 0, 0};
    v8i gz8 = {0, 0, 0, 0, 0, 0, 0, 0};
    __builtin_amdgcn_tensor_load_to_lds(g0, g1, gz4, gz4, gz8, 0);
}
#endif

// ---------------------------------------------------------------------------
// Elementwise fp32 -> bf16 convert
// ---------------------------------------------------------------------------
__global__ __launch_bounds__(256) void k_convert(const float* __restrict__ src,
                                                 __bf16* __restrict__ dst, int n) {
    int i = blockIdx.x * blockDim.x + threadIdx.x;
    if (i < n) dst[i] = f2bf(src[i]);
}

// Batched transpose + convert: dst[bb][c][r] = src[bb][r][c]
__global__ __launch_bounds__(256) void k_transpose(const float* __restrict__ src,
                                                   __bf16* __restrict__ dst,
                                                   int R, int C) {
    size_t base = (size_t)blockIdx.y * R * C;
    int i = blockIdx.x * blockDim.x + threadIdx.x;
    if (i < R * C) {
        int r = i / C, c = i % C;
        dst[base + (size_t)c * R + r] = f2bf(src[base + i]);
    }
}

// ---------------------------------------------------------------------------
// 16x64 WMMA tile (4 accumulators; A fragment reused across 4 B columns):
//   acc[g] = A[row0:+16, :K] * Bt[col0+16g : +16, :K]^T
// All 5 fragments live in distinct registers so the loads of one K-step issue
// as one clause and overlap the previous step's WMMAs (partial loadcnt waits).
// ---------------------------------------------------------------------------
__device__ inline void wmma_gemm_tile4(const __bf16* __restrict__ A, int lda, int row0,
                                       const __bf16* __restrict__ Bt, int ldb, int col0,
                                       int K, v8f acc[4]) {
    const int lane = threadIdx.x & 31;
    const int l16  = lane & 15;
    const int ah   = (lane >= 16) ? 8  : 0;
    const int bh   = (lane >= 16) ? 16 : 0;
    const __bf16* arow = A  + (size_t)(row0 + l16) * lda;
    const __bf16* bcol = Bt + (size_t)(col0 + l16) * ldb;
#pragma unroll 2
    for (int k0 = 0; k0 < K; k0 += 32) {
        frag16 a, b0, b1, b2, b3;
        a.u[0]  = *(const uint4*)(arow + k0 + bh);
        a.u[1]  = *(const uint4*)(arow + k0 + bh + 8);
        b0.u[0] = *(const uint4*)(bcol + k0 + bh);
        b0.u[1] = *(const uint4*)(bcol + k0 + bh + 8);
        b1.u[0] = *(const uint4*)(bcol + (size_t)16 * ldb + k0 + bh);
        b1.u[1] = *(const uint4*)(bcol + (size_t)16 * ldb + k0 + bh + 8);
        b2.u[0] = *(const uint4*)(bcol + (size_t)32 * ldb + k0 + bh);
        b2.u[1] = *(const uint4*)(bcol + (size_t)32 * ldb + k0 + bh + 8);
        b3.u[0] = *(const uint4*)(bcol + (size_t)48 * ldb + k0 + bh);
        b3.u[1] = *(const uint4*)(bcol + (size_t)48 * ldb + k0 + bh + 8);
        __builtin_prefetch(arow + k0 + 256, 0, 3);   // global_prefetch_b8 (near)
        acc[0] = wmma_bf16(a.v, b0.v, acc[0]);
        acc[1] = wmma_bf16(a.v, b1.v, acc[1]);
        acc[2] = wmma_bf16(a.v, b2.v, acc[2]);
        acc[3] = wmma_bf16(a.v, b3.v, acc[3]);
    }
}

// ---------------------------------------------------------------------------
// Q projection: Q[h][4096][64] = xbf @ Wqt[h]^T * SCALE   (one 16x64 tile/wave)
// ---------------------------------------------------------------------------
__global__ __launch_bounds__(32) void k_proj_q(const __bf16* __restrict__ xbf,
                                               const __bf16* __restrict__ wqt,
                                               __bf16* __restrict__ q) {
    const int mt = blockIdx.x, h = blockIdx.y;
    v8f acc[4] = {{}, {}, {}, {}};
    wmma_gemm_tile4(xbf, DIM, mt * 16, wqt + (size_t)h * DH * DIM, DIM, 0, DIM, acc);
    const int lane = threadIdx.x & 31;
    const int l16 = lane & 15, rb = (lane >= 16) ? 8 : 0;
    __bf16* outp = q + (size_t)h * MROWS * DH;
#pragma unroll
    for (int g = 0; g < 4; ++g)
#pragma unroll
        for (int r = 0; r < 8; ++r) {
            int row = mt * 16 + r + rb;
            outp[(size_t)row * DH + g * 16 + l16] = f2bf(acc[g][r] * SCALE);
        }
}

// ---------------------------------------------------------------------------
// KV projection: nt=0 -> K[4096][64]; nt=1 -> Vt[64][4096] (transposed store)
// ---------------------------------------------------------------------------
__global__ __launch_bounds__(32) void k_proj_kv(const __bf16* __restrict__ xbf,
                                                const __bf16* __restrict__ wkvt,
                                                __bf16* __restrict__ kbf,
                                                __bf16* __restrict__ vt) {
    const int mt = blockIdx.x, nt = blockIdx.y;   // nt: 0..1
    v8f acc[4] = {{}, {}, {}, {}};
    wmma_gemm_tile4(xbf, DIM, mt * 16, wkvt, DIM, nt * 64, DIM, acc);
    const int lane = threadIdx.x & 31;
    const int l16 = lane & 15, rb = (lane >= 16) ? 8 : 0;
#pragma unroll
    for (int g = 0; g < 4; ++g)
#pragma unroll
        for (int r = 0; r < 8; ++r) {
            int row = mt * 16 + r + rb;
            int col = g * 16 + l16;               // 0..63 within K or V half
            if (nt == 0)
                kbf[(size_t)row * DH + col] = f2bf(acc[g][r]);
            else
                vt[(size_t)col * MROWS + row] = f2bf(acc[g][r]);
        }
}

// ---------------------------------------------------------------------------
// Flash attention: 4 waves/block, 64 query rows; K/V block staged in LDS (TDM)
// ---------------------------------------------------------------------------
__global__ __launch_bounds__(128) void k_attn(const __bf16* __restrict__ q,
                                              const __bf16* __restrict__ kbf,
                                              const __bf16* __restrict__ vt,
                                              __bf16* __restrict__ aout) {
    __shared__ __bf16 sK[32 * DH];        // 32 keys x 64  (key-major)   4 KB
    __shared__ __bf16 sV[DH * 32];        // 64 dh   x 32 keys           4 KB
    __shared__ __bf16 sP[4 * 16 * 32];    // per-wave P tile relayout    4 KB

    const int it = blockIdx.x;            // 64-row tile in sequence
    const int h  = blockIdx.y;
    const int b  = blockIdx.z;
    const int tid = threadIdx.x;
    const int wv  = tid >> 5;
    const int lane = tid & 31;
    const int l16 = lane & 15;
    const bool hiL = lane >= 16;
    const int ah = hiL ? 8 : 0, bh = hiL ? 16 : 0, rb = hiL ? 8 : 0;
    const int row0 = b * NSEQ + it * 64 + wv * 16;   // this wave's 16 rows
    __bf16* myP = sP + wv * 16 * 32;

    // Persistent Q A-fragments (16 rows x 64, two K=32 chunks)
    const __bf16* qrow = q + ((size_t)h * MROWS + row0 + l16) * DH;
    frag16 qa[2];
#pragma unroll
    for (int kk = 0; kk < 2; ++kk) {
        qa[kk].u[0] = *(const uint4*)(qrow + kk * 32 + bh);
        qa[kk].u[1] = *(const uint4*)(qrow + kk * 32 + bh + 8);
    }

    float m[8], l[8];
    v8f acc[4] = {{}, {}, {}, {}};
#pragma unroll
    for (int r = 0; r < 8; ++r) { m[r] = -1e30f; l[r] = 0.0f; }

    for (int j0 = 0; j0 < NSEQ; j0 += 32) {
        __syncthreads();                  // previous block fully consumed
        // ---- stage K (32x64) and V (64x32) blocks into LDS -------------
#ifdef HAVE_TDM
        if (wv == 0) {
            // K block: contiguous rows, tensor/tile 64x32, stride 64
            tdm_load_2d(kbf + (size_t)(b * NSEQ + j0) * DH, sK, DH, 32, DH, 32, DH);
            // V block: 32 keys per dh-row, 64 rows, stride 4096
            tdm_load_2d(vt + (size_t)b * NSEQ + j0, sV, 32, DH, 32, DH, MROWS);
            __builtin_amdgcn_s_wait_tensorcnt(0);
        }
#else
        for (int c = tid; c < 256; c += 128) {
            ((uint4*)sK)[c] = *(const uint4*)(kbf + (size_t)(b * NSEQ + j0) * DH + c * 8);
            ((uint4*)sV)[c] = *(const uint4*)(vt + (size_t)(c >> 2) * MROWS
                                              + b * NSEQ + j0 + (c & 3) * 8);
        }
#endif
        __syncthreads();

        // ---- S = Q K^T for 32 keys: two 16x16 tiles (K frags from LDS) --
        v8f s[2];
#pragma unroll
        for (int t = 0; t < 2; ++t) {
            const __bf16* kcol = sK + (size_t)(t * 16 + l16) * DH;
            frag16 bk0, bk1;
            bk0.u[0] = *(const uint4*)(kcol + bh);
            bk0.u[1] = *(const uint4*)(kcol + bh + 8);
            bk1.u[0] = *(const uint4*)(kcol + 32 + bh);
            bk1.u[1] = *(const uint4*)(kcol + 32 + bh + 8);
            v8f c = {};
            c = wmma_bf16(qa[0].v, bk0.v, c);
            c = wmma_bf16(qa[1].v, bk1.v, c);
            s[t] = c;
        }

        // ---- online softmax (row stats via 16-lane xor reductions) ------
        float scl[8];
#pragma unroll
        for (int r = 0; r < 8; ++r) {
            float mx = fmaxf(s[0][r], s[1][r]);
            mx = fmaxf(mx, __shfl_xor(mx, 1));
            mx = fmaxf(mx, __shfl_xor(mx, 2));
            mx = fmaxf(mx, __shfl_xor(mx, 4));
            mx = fmaxf(mx, __shfl_xor(mx, 8));
            float mn = fmaxf(m[r], mx);
            scl[r] = __expf(m[r] - mn);
            m[r] = mn;
            float p0 = __expf(s[0][r] - mn);
            float p1 = __expf(s[1][r] - mn);
            s[0][r] = p0; s[1][r] = p1;
            float rs = p0 + p1;
            rs += __shfl_xor(rs, 1);
            rs += __shfl_xor(rs, 2);
            rs += __shfl_xor(rs, 4);
            rs += __shfl_xor(rs, 8);
            l[r] = l[r] * scl[r] + rs;
        }
#pragma unroll
        for (int g = 0; g < 4; ++g)
#pragma unroll
            for (int r = 0; r < 8; ++r) acc[g][r] *= scl[r];

        // ---- relayout P (C-layout) -> A-fragment via per-wave LDS tile --
#pragma unroll
        for (int t = 0; t < 2; ++t)
#pragma unroll
            for (int r = 0; r < 8; ++r)
                myP[(r + rb) * 32 + t * 16 + l16] = f2bf(s[t][r]);
        // DS ops are in-order within a wave; per-wave region needs no barrier
        frag16 pa;
        pa.u[0] = *(const uint4*)(&myP[l16 * 32 + bh]);
        pa.u[1] = *(const uint4*)(&myP[l16 * 32 + bh + 8]);

        // ---- acc += P @ V (V fragments from LDS, dh-major) --------------
#pragma unroll
        for (int g = 0; g < 4; ++g) {
            const __bf16* vcol = sV + (size_t)(g * 16 + l16) * 32;
            frag16 bf;
            bf.u[0] = *(const uint4*)(vcol + bh);
            bf.u[1] = *(const uint4*)(vcol + bh + 8);
            acc[g] = wmma_bf16(pa.v, bf.v, acc[g]);
        }
    }

    // ---- epilogue: normalize, write [4096][1024] with head interleave ---
#pragma unroll
    for (int g = 0; g < 4; ++g)
#pragma unroll
        for (int r = 0; r < 8; ++r) {
            int row = row0 + r + rb;
            int col = h * DH + g * 16 + l16;
            aout[(size_t)row * DIM + col] = f2bf(acc[g][r] / l[r]);
        }
}

// ---------------------------------------------------------------------------
// Output projection: out[4096][1024] = aout @ Woutt^T  (16x64 tile per wave)
// ---------------------------------------------------------------------------
__global__ __launch_bounds__(32) void k_proj_out(const __bf16* __restrict__ abf,
                                                 const __bf16* __restrict__ woutt,
                                                 float* __restrict__ out) {
    const int mt = blockIdx.x, nt = blockIdx.y;   // nt: 0..15 (64-col groups)
    v8f acc[4] = {{}, {}, {}, {}};
    wmma_gemm_tile4(abf, DIM, mt * 16, woutt, DIM, nt * 64, DIM, acc);
    const int lane = threadIdx.x & 31;
    const int l16 = lane & 15, rb = (lane >= 16) ? 8 : 0;
#pragma unroll
    for (int g = 0; g < 4; ++g)
#pragma unroll
        for (int r = 0; r < 8; ++r) {
            int row = mt * 16 + r + rb;
            out[(size_t)row * DIM + nt * 64 + g * 16 + l16] = acc[g][r];
        }
}

// ---------------------------------------------------------------------------
// Launch
// ---------------------------------------------------------------------------
extern "C" void kernel_launch(void* const* d_in, const int* in_sizes, int n_in,
                              void* d_out, int out_size, void* d_ws, size_t ws_size,
                              hipStream_t stream) {
    const float* x    = (const float*)d_in[0];   // [2,2048,1024]
    const float* Wq   = (const float*)d_in[1];   // [16,1024,64]
    const float* Wkv  = (const float*)d_in[2];   // [1024,128]
    const float* Wout = (const float*)d_in[3];   // [1024,1024]
    float* out = (float*)d_out;                  // [2,2048,1024]

    char* ws = (char*)d_ws;
    size_t off = 0;
    auto alloc = [&](size_t bytes) { char* p = ws + off; off += (bytes + 255) & ~(size_t)255; return p; };
    __bf16* xbf   = (__bf16*)alloc((size_t)MROWS * DIM * 2);        // 8 MB
    __bf16* wqt   = (__bf16*)alloc((size_t)HEADS * DH * DIM * 2);   // 2 MB
    __bf16* wkvt  = (__bf16*)alloc((size_t)2 * DH * DIM * 2);       // 256 KB
    __bf16* woutt = (__bf16*)alloc((size_t)DIM * DIM * 2);          // 2 MB
    __bf16* qbf   = (__bf16*)alloc((size_t)HEADS * MROWS * DH * 2); // 8 MB
    __bf16* kbf   = (__bf16*)alloc((size_t)MROWS * DH * 2);         // 512 KB
    __bf16* vtb   = (__bf16*)alloc((size_t)DH * MROWS * 2);         // 512 KB
    __bf16* aout  = (__bf16*)alloc((size_t)MROWS * DIM * 2);        // 8 MB

    // 1) precision conversion + weight transposes (bf16, B stored K-contiguous)
    int nx = MROWS * DIM;
    k_convert<<<dim3((nx + 255) / 256), 256, 0, stream>>>(x, xbf, nx);
    k_transpose<<<dim3((DIM * DH + 255) / 256, HEADS), 256, 0, stream>>>(Wq, wqt, DIM, DH);
    k_transpose<<<dim3((DIM * 2 * DH + 255) / 256, 1), 256, 0, stream>>>(Wkv, wkvt, DIM, 2 * DH);
    k_transpose<<<dim3((DIM * DIM + 255) / 256, 1), 256, 0, stream>>>(Wout, woutt, DIM, DIM);

    // 2) projections (WMMA, 16x64 tiles)
    k_proj_q<<<dim3(MROWS / 16, HEADS), 32, 0, stream>>>(xbf, wqt, qbf);
    k_proj_kv<<<dim3(MROWS / 16, 2), 32, 0, stream>>>(xbf, wkvt, kbf, vtb);

    // 3) flash attention (WMMA + TDM-staged K/V, online softmax)
    k_attn<<<dim3(NSEQ / 64, HEADS, NB), 128, 0, stream>>>(qbf, kbf, vtb, aout);

    // 4) output projection (WMMA, fp32 out)
    k_proj_out<<<dim3(MROWS / 16, DIM / 16 / 4), 32, 0, stream>>>(aout, woutt, out);

    (void)in_sizes; (void)n_in; (void)out_size; (void)ws_size;
}
